// MultiDilatelocalAttention_41162966565160
// MI455X (gfx1250) — matrix operand order repaired and checked by reading, not compiled, not run on hardware
//
#include <hip/hip_runtime.h>
#include <hip/hip_bf16.h>

// ---- problem constants (from reference) ----
#define DIM   128
#define CCH   256            // fused channels C
#define HD    32             // head dim
#define HPD   4              // heads per dilation
#define ND    2
#define BB    4
#define HH_   80
#define WW_   80
#define NPIX  (HH_ * WW_)    // 6400
#define QKVO  768            // 3*C
#define SCALE 0.17677669529663687f   // 32^-0.5

typedef __attribute__((ext_vector_type(16))) __bf16 v16bf;
typedef __attribute__((ext_vector_type(8)))  float  v8f;

union Frag16 { v16bf v; __bf16 h[16]; uint4 q[2]; };

// ---------------------------------------------------------------------------
// Kernel 0: one-shot fp32 -> bf16 conversion of both weight matrices so the
// GEMM inner loops feed WMMA straight from b128 loads (no v_cvt, no WAR NOPs).
// ---------------------------------------------------------------------------
__global__ __launch_bounds__(256)
void cvt_weights_kernel(const float* __restrict__ wqkv, const float* __restrict__ wproj,
                        __bf16* __restrict__ wqkv_bf, __bf16* __restrict__ wproj_bf)
{
    const int i = blockIdx.x * 256 + threadIdx.x;
    if (i < QKVO * CCH) wqkv_bf[i] = (__bf16)wqkv[i];
    if (i < CCH * CCH)  wproj_bf[i] = (__bf16)wproj[i];
}

// ---------------------------------------------------------------------------
// Kernel 1: QKV projection.  GEMM [25600 x 256] x [256 x 768].
// Block = 64 positions x 256 out-channels; 8 waves, each wave: 4 M-subtiles
// x 2 N-tiles (8 accumulators).  B fragments reused across 4 A fragments.
// ---------------------------------------------------------------------------
__global__ __launch_bounds__(256)
void qkv_gemm_kernel(const float* __restrict__ rgb, const float* __restrict__ ir,
                     const __bf16* __restrict__ wbf, float* __restrict__ qkv)
{
    __shared__ __align__(16) __bf16 ylds[64 * CCH];   // [pos][channel], 32 KB
    const int tid     = threadIdx.x;
    const int rowbase = blockIdx.x * 64;              // global row = b*NPIX + n
    const int b       = rowbase / NPIX;
    const int n0      = rowbase % NPIX;

    // stage y tile (64 pos x 256 ch) into LDS as bf16; coalesced in p
    for (int idx = tid; idx < 64 * CCH; idx += 256) {
        const int p = idx & 63;
        const int c = idx >> 6;
        const float* src = (c < DIM) ? rgb : ir;
        ylds[p * CCH + c] =
            (__bf16)src[((size_t)b * DIM + (c & (DIM - 1))) * NPIX + n0 + p];
    }
    __syncthreads();

    const int wave = tid >> 5;
    const int lane = tid & 31;
    const int col  = lane & 15;
    const int hf   = lane >> 4;
    const int m0   = hf * 8;
    const int obase = blockIdx.y * 256 + wave * 32;   // this wave's 32 channels

    v8f acc[4][2] = {};

    for (int kk = 0; kk < 8; ++kk) {
        const int kbase = kk * 32;
        const int kb    = kbase + hf * 16;            // lanes>=16 hold K=16..31

        // two B fragments (32x16 bf16 each), 16B vector loads, bf16 direct
        Frag16 b0, b1;
        const uint4* wp0 = reinterpret_cast<const uint4*>(wbf + (size_t)(obase + col) * CCH + kb);
        const uint4* wp1 = reinterpret_cast<const uint4*>(wbf + (size_t)(obase + 16 + col) * CCH + kb);
        b0.q[0] = wp0[0]; b0.q[1] = wp0[1];
        b1.q[0] = wp1[0]; b1.q[1] = wp1[1];

        const int alo = kbase + hf * 16;
        #pragma unroll
        for (int mt = 0; mt < 4; ++mt) {
            Frag16 a;
            const int prow = mt * 16 + col;
            a.q[0] = *reinterpret_cast<const uint4*>(&ylds[prow * CCH + alo]);
            a.q[1] = *reinterpret_cast<const uint4*>(&ylds[prow * CCH + alo + 8]);
            acc[mt][0] = __builtin_amdgcn_wmma_f32_16x16x32_bf16(
                false, a.v, false, b0.v, (short)0, acc[mt][0], false, false);
            acc[mt][1] = __builtin_amdgcn_wmma_f32_16x16x32_bf16(
                false, a.v, false, b1.v, (short)0, acc[mt][1], false, false);
        }
    }

    // D layout: VGPR r -> M=r (lanes 0-15) / r+8 (lanes 16-31); N = col.
    #pragma unroll
    for (int mt = 0; mt < 4; ++mt) {
        #pragma unroll
        for (int t = 0; t < 2; ++t) {
            const int o = obase + t * 16 + col;
            float* dst = qkv + ((size_t)b * QKVO + o) * NPIX + n0 + mt * 16 + m0;
            *reinterpret_cast<float4*>(dst) =
                make_float4(acc[mt][t][0], acc[mt][t][1], acc[mt][t][2], acc[mt][t][3]);
            *reinterpret_cast<float4*>(dst + 4) =
                make_float4(acc[mt][t][4], acc[mt][t][5], acc[mt][t][6], acc[mt][t][7]);
        }
    }
}

// ---------------------------------------------------------------------------
// Kernel 2: dilated 3x3 local attention (softmax over 9 incl. zero padding).
// One thread per (b, dilation, head, pixel); consecutive threads = consecutive
// pixels -> coalesced channel-major reads; k/v reuse served by 192MB L2.
// ---------------------------------------------------------------------------
__global__ __launch_bounds__(256)
void attn_kernel(const float* __restrict__ qkv, __bf16* __restrict__ xbuf)
{
    const int g  = blockIdx.x * 256 + threadIdx.x;   // 204800 total
    const int n  = g % NPIX;
    int r        = g / NPIX;
    const int hh = r % HPD;  r /= HPD;
    const int di = r % ND;
    const int b  = r / ND;
    const int py = n / WW_, px = n % WW_;
    const int d  = (di == 0) ? 2 : 3;

    const size_t chbase = (size_t)b * QKVO;
    const int qc0 = di * 128 + hh * HD;
    const float* qptr = qkv + (chbase + qc0) * NPIX + n;
    const float* kptr = qkv + (chbase + 256 + qc0) * NPIX;
    const float* vptr = qkv + (chbase + 512 + qc0) * NPIX;

    float q[HD];
    #pragma unroll
    for (int c = 0; c < HD; ++c) q[c] = qptr[(size_t)c * NPIX];

    float logit[9];
    int   nn[9];
    #pragma unroll
    for (int ki = 0; ki < 3; ++ki) {
        #pragma unroll
        for (int kj = 0; kj < 3; ++kj) {
            const int idx = ki * 3 + kj;
            const int hy = py + (ki - 1) * d;
            const int wx = px + (kj - 1) * d;
            const bool ok = (hy >= 0) && (hy < HH_) && (wx >= 0) && (wx < WW_);
            const int np = hy * WW_ + wx;
            nn[idx] = ok ? np : -1;
            float dot = 0.f;
            if (ok) {
                #pragma unroll
                for (int c = 0; c < HD; ++c)
                    dot += q[c] * kptr[(size_t)c * NPIX + np];
            }
            logit[idx] = dot * SCALE;   // padded neighbor stays in softmax with logit 0
        }
    }

    float m = logit[0];
    #pragma unroll
    for (int i = 1; i < 9; ++i) m = fmaxf(m, logit[i]);
    float p[9], s = 0.f;
    #pragma unroll
    for (int i = 0; i < 9; ++i) { p[i] = __expf(logit[i] - m); s += p[i]; }
    const float inv = 1.f / s;

    float out[HD];
    #pragma unroll
    for (int c = 0; c < HD; ++c) out[c] = 0.f;
    #pragma unroll
    for (int i = 0; i < 9; ++i) {
        if (nn[i] >= 0) {
            const float wgt = p[i] * inv;
            #pragma unroll
            for (int c = 0; c < HD; ++c)
                out[c] += wgt * vptr[(size_t)c * NPIX + nn[i]];
        }
    }

    __bf16* dst = xbuf + ((size_t)b * NPIX + n) * CCH + di * 128 + hh * HD;
    #pragma unroll
    for (int c = 0; c < HD; ++c) dst[c] = (__bf16)out[c];
}

// ---------------------------------------------------------------------------
// Kernel 3: output projection + bias + residual.  GEMM [25600x256]x[256x256].
// Block = 64 positions x 256 channels; same 4 M-subtiles x 2 N-tiles per wave.
// ---------------------------------------------------------------------------
__global__ __launch_bounds__(256)
void proj_kernel(const __bf16* __restrict__ xbuf, const __bf16* __restrict__ wbf,
                 const float* __restrict__ b_proj, const float* __restrict__ rgb,
                 const float* __restrict__ ir, float* __restrict__ out)
{
    const int tid     = threadIdx.x;
    const int rowbase = blockIdx.x * 64;
    const int b       = rowbase / NPIX;
    const int n0      = rowbase % NPIX;
    const int wave    = tid >> 5;
    const int lane    = tid & 31;
    const int col     = lane & 15;
    const int hf      = lane >> 4;
    const int m0      = hf * 8;
    const int obase   = wave * 32;

    v8f acc[4][2] = {};

    for (int kk = 0; kk < 8; ++kk) {
        const int kbase = kk * 32;
        const int kb    = kbase + hf * 16;

        Frag16 b0, b1;
        const uint4* wp0 = reinterpret_cast<const uint4*>(wbf + (size_t)(obase + col) * CCH + kb);
        const uint4* wp1 = reinterpret_cast<const uint4*>(wbf + (size_t)(obase + 16 + col) * CCH + kb);
        b0.q[0] = wp0[0]; b0.q[1] = wp0[1];
        b1.q[0] = wp1[0]; b1.q[1] = wp1[1];

        const int alo = kbase + hf * 16;
        #pragma unroll
        for (int mt = 0; mt < 4; ++mt) {
            Frag16 a;
            const __bf16* arow = xbuf + ((size_t)rowbase + mt * 16 + col) * CCH;
            a.q[0] = *reinterpret_cast<const uint4*>(arow + alo);
            a.q[1] = *reinterpret_cast<const uint4*>(arow + alo + 8);
            acc[mt][0] = __builtin_amdgcn_wmma_f32_16x16x32_bf16(
                false, a.v, false, b0.v, (short)0, acc[mt][0], false, false);
            acc[mt][1] = __builtin_amdgcn_wmma_f32_16x16x32_bf16(
                false, a.v, false, b1.v, (short)0, acc[mt][1], false, false);
        }
    }

    #pragma unroll
    for (int mt = 0; mt < 4; ++mt) {
        #pragma unroll
        for (int t = 0; t < 2; ++t) {
            const int o = obase + t * 16 + col;
            const float bias = b_proj[o];
            const float* res = (o < DIM) ? rgb : ir;
            const size_t resbase =
                ((size_t)b * DIM + (o & (DIM - 1))) * NPIX + n0 + mt * 16 + m0;
            #pragma unroll
            for (int rr = 0; rr < 8; ++rr) {
                const int row = rowbase + mt * 16 + m0 + rr;
                out[(size_t)row * CCH + o] = acc[mt][t][rr] + bias + res[resbase + rr];
            }
        }
    }
}

// ---------------------------------------------------------------------------
extern "C" void kernel_launch(void* const* d_in, const int* in_sizes, int n_in,
                              void* d_out, int out_size, void* d_ws, size_t ws_size,
                              hipStream_t stream) {
    const float* rgb    = (const float*)d_in[0];
    const float* ir     = (const float*)d_in[1];
    const float* w_qkv  = (const float*)d_in[2];
    const float* w_proj = (const float*)d_in[3];
    const float* b_proj = (const float*)d_in[4];
    float* out = (float*)d_out;

    char* ws = (char*)d_ws;
    size_t off = 0;
    float*  qkv     = (float*)(ws + off);  off += (size_t)BB * QKVO * NPIX * 4;  // 78.6 MB
    __bf16* xbuf    = (__bf16*)(ws + off); off += (size_t)BB * NPIX * CCH * 2;   // 13.1 MB
    __bf16* wqkv_bf = (__bf16*)(ws + off); off += (size_t)QKVO * CCH * 2;        // 384 KB
    __bf16* wproj_bf= (__bf16*)(ws + off);                                       // 128 KB

    cvt_weights_kernel<<<dim3((QKVO * CCH) / 256), dim3(256), 0, stream>>>(
        w_qkv, w_proj, wqkv_bf, wproj_bf);

    qkv_gemm_kernel<<<dim3(BB * NPIX / 64, 3), dim3(256), 0, stream>>>(
        rgb, ir, wqkv_bf, qkv);

    attn_kernel<<<dim3((BB * ND * HPD * NPIX) / 256), dim3(256), 0, stream>>>(qkv, xbuf);

    proj_kernel<<<dim3(BB * NPIX / 64), dim3(256), 0, stream>>>(
        xbuf, wproj_bf, b_proj, rgb, ir, out);
}
